// AllegroScalarOutputHead_14405320311651
// MI455X (gfx1250) — matrix operand order repaired and checked by reading, not compiled, not run on hardware
//
#include <hip/hip_runtime.h>
#include <hip/hip_bf16.h>

#define MAXZ1 101   // MAX_Z + 1

typedef __attribute__((ext_vector_type(16))) __bf16 bf16x16;
typedef __attribute__((ext_vector_type(16))) float  floatx16;
typedef __attribute__((ext_vector_type(8)))  float  floatx8;
typedef __attribute__((ext_vector_type(4)))  float  fx4;

union Frag {
    bf16x16 v;
    uint4   q[2];
};

// Computes, for a 128-row strip starting at row0, the per-row scalar
//   part -> sum_n silu(X[m,:] @ W1[:,n] + b1[n]) * W2[n]
// using v_wmma_f32_16x16x32_bf16. 256 threads = 8 waves, each wave owns 16 rows.
// After return: lanes 0-15 hold the sum for row m0+r, lanes 16-31 for row m0+8+r.
template<int D, int NG>
__device__ __forceinline__ void mlp_row_head(
    const float* __restrict__ X, int M, int row0,
    const float* __restrict__ W1, const float* __restrict__ b1,
    const float* __restrict__ W2, float part[8])
{
    constexpr int PAD    = 8;        // bf16 elements of padding per LDS column
    constexpr int KSTEPS = D / 32;
    extern __shared__ __bf16 Blds[];   // [NG][D+PAD] bf16, column-major weights

    const int tid  = threadIdx.x;
    const int wave = tid >> 5;
    const int lane = tid & 31;
    const int lh   = lane & 15;
    const int hi   = lane >> 4;      // 0: lanes 0-15, 1: lanes 16-31

    const int m0 = row0 + wave * 16;
    int arow = m0 + lh;
    if (arow >= M) arow = M - 1;     // clamp loads; writes are guarded by caller
    const float* rowp = X + (size_t)arow * D;
    __builtin_prefetch(rowp, 0, 3);  // global_prefetch_b8: warm the row while staging B

    // A fragments, contiguous k order (B uses the same): lane half 0 holds K {0..15},
    // half 1 holds K {16..31} of each 32-wide K step.
    // Features are streamed exactly once -> non-temporal loads keep L2 free for
    // the weight matrices / lookup tables / accumulators that ARE reused.
    Frag a[KSTEPS];
#pragma unroll
    for (int ks = 0; ks < KSTEPS; ++ks) {
        const float* p = rowp + ks * 32 + hi * 16;
        fx4 f0 = __builtin_nontemporal_load((const fx4*)(p));
        fx4 f1 = __builtin_nontemporal_load((const fx4*)(p + 4));
        fx4 f2 = __builtin_nontemporal_load((const fx4*)(p + 8));
        fx4 f3 = __builtin_nontemporal_load((const fx4*)(p + 12));
        floatx16 t;
        t[0]  = f0.x; t[1]  = f0.y; t[2]  = f0.z; t[3]  = f0.w;
        t[4]  = f1.x; t[5]  = f1.y; t[6]  = f1.z; t[7]  = f1.w;
        t[8]  = f2.x; t[9]  = f2.y; t[10] = f2.z; t[11] = f2.w;
        t[12] = f3.x; t[13] = f3.y; t[14] = f3.z; t[15] = f3.w;
        a[ks].v = __builtin_convertvector(t, bf16x16);  // packed v_cvt bf16 converts
    }

#pragma unroll
    for (int r = 0; r < 8; ++r) part[r] = 0.0f;

    // Loop over output-column groups; stage W1 columns transposed as bf16 in LDS.
    // float4-vectorized along n (contiguous in row-major W1); W1 stays L2-resident.
    for (int n0 = 0; n0 < D; n0 += NG) {
        __syncthreads();
        for (int i = tid; i < (NG / 4) * D; i += 256) {
            int nl4 = (i % (NG / 4)) * 4;     // consecutive tid -> consecutive global reads
            int k   = i / (NG / 4);
            fx4 w = *(const fx4*)(W1 + (size_t)k * D + n0 + nl4);
            Blds[(nl4 + 0) * (D + PAD) + k] = (__bf16)w.x;
            Blds[(nl4 + 1) * (D + PAD) + k] = (__bf16)w.y;
            Blds[(nl4 + 2) * (D + PAD) + k] = (__bf16)w.z;
            Blds[(nl4 + 3) * (D + PAD) + k] = (__bf16)w.w;
        }
        __syncthreads();

#pragma unroll
        for (int nt = 0; nt < NG / 16; ++nt) {
            const int nloc = nt * 16 + lh;
            floatx8 acc = {0.f, 0.f, 0.f, 0.f, 0.f, 0.f, 0.f, 0.f};
            // B in the same contiguous k order as A: lanes 0-15 hold K 0..15 of column n, lanes 16-31 hold K 16..31.
            const __bf16* bbase = Blds + nloc * (D + PAD) + hi * 16;
#pragma unroll
            for (int ks = 0; ks < KSTEPS; ++ks) {
                Frag b;
                const uint4* bp = (const uint4*)(bbase + ks * 32);  // 16B-aligned by PAD choice
                b.q[0] = bp[0];
                b.q[1] = bp[1];
                acc = __builtin_amdgcn_wmma_f32_16x16x32_bf16(
                    false, a[ks].v, false, b.v, (short)0, acc, false, false);
            }
            // Fused epilogue: silu(h + b1) * W2, accumulated per C-matrix row.
            const int   ncol = n0 + nt * 16 + lh;
            const float b1n  = b1[ncol];
            const float w2n  = W2[ncol];
#pragma unroll
            for (int r = 0; r < 8; ++r) {
                float hx = acc[r] + b1n;
                float s  = hx * (1.0f / (1.0f + __expf(-hx)));
                part[r] += s * w2n;
            }
        }
    }

    // Reduce across N within each 16-lane half (C layout: VGPR r = row m0+r (lanes 0-15)
    // and row m0+8+r (lanes 16-31)).
#pragma unroll
    for (int r = 0; r < 8; ++r) {
#pragma unroll
        for (int off = 1; off < 16; off <<= 1)
            part[r] += __shfl_xor(part[r], off, 32);
    }
}

__global__ void __launch_bounds__(256)
zero_kernel(float* __restrict__ node_accum, int n, float* __restrict__ out, int m)
{
    int i = blockIdx.x * blockDim.x + threadIdx.x;
    if (i < n) node_accum[i] = 0.0f;
    if (i < m) out[i] = 0.0f;
}

__global__ void __launch_bounds__(256)
edge_kernel(const float* __restrict__ Xe,
            const float* __restrict__ W1e, const float* __restrict__ b1e,
            const float* __restrict__ W2e, const float* __restrict__ b2e,
            const float* __restrict__ pair_scales,
            const int* __restrict__ z,
            const int* __restrict__ idx_s, const int* __restrict__ idx_t,
            float* __restrict__ node_accum, int M)
{
    float part[8];
    mlp_row_head<128, 128>(Xe, M, blockIdx.x * 128, W1e, b1e, W2e, part);

    const int lh   = threadIdx.x & 15;
    const int hi   = (threadIdx.x >> 4) & 1;
    const int wave = threadIdx.x >> 5;
    const int m0   = blockIdx.x * 128 + wave * 16;
    const float b2 = b2e[0];

#pragma unroll
    for (int r = 0; r < 8; ++r) {
        if (lh == r) {
            int e = m0 + r + 8 * hi;
            if (e < M) {
                float pe = part[r] + b2;
                int it = idx_t[e];
                int zs = z[idx_s[e]];
                int zt = z[it];
                pe *= pair_scales[zs * MAXZ1 + zt];
                atomicAdd(&node_accum[it], pe);   // segment_sum(edges -> nodes)
            }
        }
    }
}

__global__ void __launch_bounds__(256)
node_kernel(const float* __restrict__ Xn,
            const float* __restrict__ W1n, const float* __restrict__ b1n,
            const float* __restrict__ W2n, const float* __restrict__ b2n,
            const float* __restrict__ atom_scales,
            const float* __restrict__ atom_shifts,
            const int* __restrict__ z, const int* __restrict__ batch,
            const float* __restrict__ node_accum,
            float* __restrict__ out, int M)
{
    float part[8];
    mlp_row_head<256, 64>(Xn, M, blockIdx.x * 128, W1n, b1n, W2n, part);

    const int lh   = threadIdx.x & 15;
    const int hi   = (threadIdx.x >> 4) & 1;
    const int wave = threadIdx.x >> 5;
    const int m0   = blockIdx.x * 128 + wave * 16;
    const float b2 = b2n[0];

#pragma unroll
    for (int r = 0; r < 8; ++r) {
        if (lh == r) {
            int nid = m0 + r + 8 * hi;
            if (nid < M) {
                int   zi = z[nid];
                float pa = part[r] + b2 + node_accum[nid];
                pa = pa * atom_scales[zi] + atom_shifts[zi];
                atomicAdd(&out[batch[nid]], pa);  // segment_sum(nodes -> graphs)
            }
        }
    }
}

extern "C" void kernel_launch(void* const* d_in, const int* in_sizes, int n_in,
                              void* d_out, int out_size, void* d_ws, size_t ws_size,
                              hipStream_t stream)
{
    const float* node_feats  = (const float*)d_in[0];
    const float* edge_feats  = (const float*)d_in[1];
    const int*   atomic_num  = (const int*)  d_in[2];
    const int*   idx_s       = (const int*)  d_in[3];
    const int*   idx_t       = (const int*)  d_in[4];
    const int*   batch       = (const int*)  d_in[5];
    const float* W1n         = (const float*)d_in[6];
    const float* b1n         = (const float*)d_in[7];
    const float* W2n         = (const float*)d_in[8];
    const float* b2n         = (const float*)d_in[9];
    const float* W1e         = (const float*)d_in[10];
    const float* b1e         = (const float*)d_in[11];
    const float* W2e         = (const float*)d_in[12];
    const float* b2e         = (const float*)d_in[13];
    const float* atom_scales = (const float*)d_in[14];
    const float* atom_shifts = (const float*)d_in[15];
    const float* pair_scales = (const float*)d_in[16];

    float* out        = (float*)d_out;
    float* node_accum = (float*)d_ws;          // N_NODES floats of scratch

    const int n_nodes = in_sizes[2];           // 50000
    const int n_edges = in_sizes[3];           // 640000

    // 1) zero accumulators (harness poisons d_ws/d_out; must re-zero every call)
    zero_kernel<<<(n_nodes + 255) / 256, 256, 0, stream>>>(node_accum, n_nodes, out, out_size);

    // 2) edge MLP + pair-scale + scatter to nodes
    {
        const int blocks = (n_edges + 127) / 128;
        const size_t smem = (size_t)128 * (128 + 8) * sizeof(__bf16); // 34 KB
        edge_kernel<<<blocks, 256, smem, stream>>>(
            edge_feats, W1e, b1e, W2e, b2e, pair_scales,
            atomic_num, idx_s, idx_t, node_accum, n_edges);
    }

    // 3) node MLP + edge accumulation + atom scale/shift + scatter to graphs
    {
        const int blocks = (n_nodes + 127) / 128;
        const size_t smem = (size_t)64 * (256 + 8) * sizeof(__bf16);  // 33 KB
        node_kernel<<<blocks, 256, smem, stream>>>(
            node_feats, W1n, b1n, W2n, b2n, atom_scales, atom_shifts,
            atomic_num, batch, node_accum, out, n_nodes);
    }
}
